// MultiModalFusionWithCMA_39178691674139
// MI455X (gfx1250) — matrix operand order repaired and checked by reading, not compiled, not run on hardware
//
#include <hip/hip_runtime.h>

typedef __attribute__((ext_vector_type(16))) _Float16 v16h;
typedef __attribute__((ext_vector_type(8)))  _Float16 v8h;
typedef __attribute__((ext_vector_type(4)))  _Float16 v4h;
typedef __attribute__((ext_vector_type(8)))  float    v8f;
typedef __attribute__((ext_vector_type(4)))  float    v4f;

#define BATCH 32
#define CDIM  768
#define HWDIM 576
#define NELEM (BATCH * CDIM * HWDIM)          // 14,155,776 elements per tensor
#define SPITCH 772                            // lds_s row pitch (f32): 772%64=4 -> conflict-free
#define PPITCH 776                            // lds_p row pitch (f16): 1552B/16 aligned, bank 4*lane

// ============================================================================
// Pre-pass 1: f32 -> f16 copies of feat1 and feat2 (channel-major)
// ============================================================================
__global__ __launch_bounds__(256)
void convert_f16_kernel(const float* __restrict__ f1, const float* __restrict__ f2,
                        _Float16* __restrict__ f1h, _Float16* __restrict__ f2h)
{
    size_t i = ((size_t)blockIdx.x * 256 + threadIdx.x) * 4;
    if (i >= (size_t)NELEM) return;
    v4f a = *(const v4f*)(f1 + i);
    v4f b = *(const v4f*)(f2 + i);
    v4h ah = { (_Float16)a[0], (_Float16)a[1], (_Float16)a[2], (_Float16)a[3] };
    v4h bh = { (_Float16)b[0], (_Float16)b[1], (_Float16)b[2], (_Float16)b[3] };
    *(v4h*)(f1h + i) = ah;
    *(v4h*)(f2h + i) = bh;
}

// ============================================================================
// Pre-pass 2: feat2 [b][d][n] f32 -> f16 transposed [b][n][d]
// ============================================================================
__global__ __launch_bounds__(256)
void transpose_f16_kernel(const float* __restrict__ f2, _Float16* __restrict__ f2t)
{
    __shared__ _Float16 tile[32][33];
    const int b  = blockIdx.z;
    const int d0 = blockIdx.x * 32;           // 24 tiles over CDIM
    const int n0 = blockIdx.y * 32;           // 18 tiles over HWDIM
    const int tx = threadIdx.x, ty = threadIdx.y;   // 32 x 8

    const float* src = f2 + ((size_t)b * CDIM + d0) * HWDIM + n0;
    #pragma unroll
    for (int r = ty; r < 32; r += 8)
        tile[r][tx] = (_Float16)src[(size_t)r * HWDIM + tx];
    __syncthreads();
    _Float16* dst = f2t + ((size_t)b * HWDIM + n0) * CDIM + d0;
    #pragma unroll
    for (int r = ty; r < 32; r += 8)
        dst[(size_t)r * CDIM + tx] = tile[tx][r];
}

// ============================================================================
// Main kernel (fast path): 256 threads = 8 wave32 waves, 32-row channel block.
// All WMMA operands come straight from global (L2-resident f16 copies);
// LDS holds only the logits strip, P, and softmax scratch.
// ============================================================================
__global__ __launch_bounds__(256, 2)
void cma_wmma_kernel(const float* __restrict__ feat1,
                     const _Float16* __restrict__ f1h,
                     const _Float16* __restrict__ f2h,
                     const _Float16* __restrict__ f2t,
                     const float* __restrict__ gamma,
                     float* __restrict__ out)
{
    __shared__ float    lds_s[32 * SPITCH];   // logits, ~98.8 KB
    __shared__ _Float16 lds_p[32 * PPITCH];   // exp(S-max), ~49.7 KB
    __shared__ float    red[256];
    __shared__ float    inv_row[32];

    const int tid   = threadIdx.x;
    const int wave  = tid >> 5;
    const int lane  = tid & 31;
    const int lhalf = (lane < 16) ? 0 : 1;
    const int l16   = lane & 15;
    const int rh    = wave & 1;               // row-half of the 32-row block
    const int cg    = wave >> 1;              // column group (0..3)

    const int batch = blockIdx.x / (CDIM / 32);
    const int m0    = (blockIdx.x % (CDIM / 32)) * 32;

    // ---- stage 1: S[32,768] = Q * K^T ------------------------------------
    // Chunk-outer / tile-inner: A (Q) fragment loaded once per K-chunk and
    // reused across all 12 d-tiles; 12 S-tiles accumulate in registers.
    {
        const _Float16* qrow = f1h + ((size_t)batch * CDIM + m0 + rh * 16 + l16) * HWDIM;
        // Base B row for this wave's first d-tile (dtile = cg): consecutive
        // tiles are 4*16 rows apart -> constant 73728B immediate offsets.
        const _Float16* krow0 = f2h + ((size_t)batch * CDIM + cg * 16 + l16) * HWDIM;

        v8f sacc[12];
        #pragma unroll
        for (int t = 0; t < 12; ++t) sacc[t] = (v8f){};

        for (int c = 0; c < 18; ++c) {        // 576/32 K-chunks
            const int k0 = c * 32;
            v8h alo = *(const v8h*)(qrow + k0 + lhalf * 16);
            v8h ahi = *(const v8h*)(qrow + k0 + lhalf * 16 + 8);
            v16h a;
            #pragma unroll
            for (int h = 0; h < 8; ++h) { a[h] = alo[h]; a[h + 8] = ahi[h]; }
            #pragma unroll
            for (int t = 0; t < 12; ++t) {
                // B: contiguous 16-K group, tile t is 64 rows (36864 halves) away
                v16h bb = *(const v16h*)(krow0 + (size_t)t * 64 * HWDIM + k0 + lhalf * 16);
                sacc[t] = __builtin_amdgcn_wmma_f32_16x16x32_f16(false, a, false, bb,
                                                                 (short)0, sacc[t], false, false);
            }
        }
        #pragma unroll
        for (int t = 0; t < 12; ++t) {
            const int scol = (cg + 4 * t) * 16 + l16;
            #pragma unroll
            for (int r = 0; r < 8; ++r)
                lds_s[(rh * 16 + r + lhalf * 8) * SPITCH + scol] = sacc[t][r];
        }
    }
    __syncthreads();

    // ---- softmax over 32 rows, 8 threads per row -------------------------
    {
        const int r = tid >> 3, sub = tid & 7;
        float m = -3.0e38f;
        for (int c = sub; c < CDIM; c += 8)
            m = fmaxf(m, lds_s[r * SPITCH + c]);
        red[tid] = m;
        __syncthreads();
        if (sub == 0) {
            float mm = red[r * 8];
            #pragma unroll
            for (int k = 1; k < 8; ++k) mm = fmaxf(mm, red[r * 8 + k]);
            red[r * 8] = mm;
        }
        __syncthreads();
        m = red[r * 8];
        __syncthreads();
        float s = 0.f;
        for (int c = sub; c < CDIM; c += 8) {
            float e = __expf(lds_s[r * SPITCH + c] - m);
            lds_p[r * PPITCH + c] = (_Float16)e;  // unnormalized
            s += e;
        }
        red[tid] = s;
        __syncthreads();
        if (sub == 0) {
            float ss = 0.f;
            #pragma unroll
            for (int k = 0; k < 8; ++k) ss += red[r * 8 + k];
            inv_row[r] = 1.0f / ss;
        }
        __syncthreads();
    }

    // ---- stage 2: O[32,576] = P * V --------------------------------------
    v8f oacc[9];
    #pragma unroll
    for (int j = 0; j < 9; ++j) oacc[j] = (v8f){};

    const _Float16* prow = lds_p + (rh * 16 + l16) * PPITCH;
    const _Float16* vrow = f2t + ((size_t)batch * HWDIM + cg * 16 + l16) * CDIM;

    for (int c2 = 0; c2 < 24; ++c2) {         // 768/32 K-chunks
        const int d0 = c2 * 32;
        v8h alo = *(const v8h*)(prow + d0 + lhalf * 16);        // ds_load_b128
        v8h ahi = *(const v8h*)(prow + d0 + lhalf * 16 + 8);
        v16h a;
        #pragma unroll
        for (int h = 0; h < 8; ++h) { a[h] = alo[h]; a[h + 8] = ahi[h]; }
        #pragma unroll
        for (int j = 0; j < 9; ++j) {          // 36 col-tiles per row-half / 4 waves
            // B[k=d][n]: lane n holds 16 consecutive d -> one 32B load;
            // col-tile j is 64 spatial rows (49152 halves) away.
            v16h bb = *(const v16h*)(vrow + (size_t)j * 64 * CDIM + d0 + lhalf * 16);
            oacc[j] = __builtin_amdgcn_wmma_f32_16x16x32_f16(false, a, false, bb,
                                                             (short)0, oacc[j], false, false);
        }
    }

    // ---- epilogue: out = gamma * O / rowsum + feat1 ----------------------
    const float g = gamma[0];
    #pragma unroll
    for (int j = 0; j < 9; ++j) {
        const int col = (cg + 4 * j) * 16 + l16;
        #pragma unroll
        for (int r = 0; r < 8; ++r) {
            const int M = rh * 16 + r + lhalf * 8;
            const size_t idx = ((size_t)batch * CDIM + m0 + M) * HWDIM + col;
            out[idx] = g * oacc[j][r] * inv_row[M] + feat1[idx];
        }
    }
}

// ============================================================================
// Fallback (no workspace): fused kernel, LDS-staged, 128 threads.
// ============================================================================
__global__ __launch_bounds__(128, 1)
void cma_fused_kernel(const float* __restrict__ feat1,
                      const float* __restrict__ feat2,
                      const float* __restrict__ gamma,
                      float* __restrict__ out)
{
    __shared__ _Float16 lds_q [16 * HWDIM];
    __shared__ _Float16 lds_kv[64 * HWDIM];
    __shared__ float    lds_s [16 * CDIM];
    __shared__ _Float16 lds_p [16 * CDIM];
    __shared__ float    red[128];
    __shared__ float    inv_row[16];

    const int tid   = threadIdx.x;
    const int wave  = tid >> 5;
    const int lane  = tid & 31;
    const int lhalf = (lane < 16) ? 0 : 1;
    const int l16   = lane & 15;

    const int batch = blockIdx.x / (CDIM / 16);
    const int m0    = (blockIdx.x % (CDIM / 16)) * 16;

    const float* f1b = feat1 + (size_t)batch * CDIM * HWDIM;
    const float* f2b = feat2 + (size_t)batch * CDIM * HWDIM;

    {
        const float* qsrc = f1b + (size_t)m0 * HWDIM;
        for (int i = tid; i < 16 * HWDIM; i += 128)
            lds_q[i] = (_Float16)qsrc[i];
    }
    __syncthreads();

    for (int sb = 0; sb < 12; ++sb) {
        const float* ksrc = f2b + (size_t)(sb * 64) * HWDIM;
        for (int i = tid; i < 64 * HWDIM; i += 128)
            lds_kv[i] = (_Float16)ksrc[i];
        __syncthreads();

        v8f acc = {};
        const int drow = wave * 16 + l16;
        for (int c = 0; c < 18; ++c) {
            const int k0 = c * 32;
            v16h a, bb;
            const int abase = l16 * HWDIM + k0 + lhalf * 16;
            #pragma unroll
            for (int h = 0; h < 8; ++h) {
                a[h]     = lds_q[abase + h];
                a[h + 8] = lds_q[abase + 8 + h];
            }
            const int bbase = drow * HWDIM + k0 + lhalf * 16;
            #pragma unroll
            for (int h = 0; h < 16; ++h)
                bb[h] = lds_kv[bbase + h];
            acc = __builtin_amdgcn_wmma_f32_16x16x32_f16(false, a, false, bb,
                                                         (short)0, acc, false, false);
        }
        const int scol = sb * 64 + wave * 16 + l16;
        #pragma unroll
        for (int r = 0; r < 8; ++r)
            lds_s[(r + lhalf * 8) * CDIM + scol] = acc[r];
        __syncthreads();
    }

    {
        const int r = tid >> 3, sub = tid & 7;
        float m = -3.0e38f;
        for (int c = sub; c < CDIM; c += 8)
            m = fmaxf(m, lds_s[r * CDIM + c]);
        red[tid] = m;
        __syncthreads();
        if (sub == 0) {
            float mm = red[r * 8];
            #pragma unroll
            for (int k = 1; k < 8; ++k) mm = fmaxf(mm, red[r * 8 + k]);
            red[r * 8] = mm;
        }
        __syncthreads();
        m = red[r * 8];
        __syncthreads();
        float s = 0.f;
        for (int c = sub; c < CDIM; c += 8) {
            float e = __expf(lds_s[r * CDIM + c] - m);
            lds_p[r * CDIM + c] = (_Float16)e;
            s += e;
        }
        red[tid] = s;
        __syncthreads();
        if (sub == 0) {
            float ss = 0.f;
            #pragma unroll
            for (int k = 0; k < 8; ++k) ss += red[r * 8 + k];
            inv_row[r] = 1.0f / ss;
        }
        __syncthreads();
    }

    v8f oacc[9];
    #pragma unroll
    for (int j = 0; j < 9; ++j) oacc[j] = (v8f){};

    for (int c2 = 0; c2 < 24; ++c2) {
        const int d0 = c2 * 32;
        const float* vsrc = f2b + (size_t)d0 * HWDIM;
        for (int i = tid; i < 32 * HWDIM; i += 128)
            lds_kv[i] = (_Float16)vsrc[i];
        __syncthreads();

        v16h a;
        const int abase = l16 * CDIM + d0 + lhalf * 16;
        #pragma unroll
        for (int h = 0; h < 8; ++h) {
            a[h]     = lds_p[abase + h];
            a[h + 8] = lds_p[abase + 8 + h];
        }
        #pragma unroll
        for (int j = 0; j < 9; ++j) {
            const int n0 = (wave + 4 * j) * 16;
            v16h bb;
            const int bbase = (lhalf * 16) * HWDIM + n0 + l16;
            #pragma unroll
            for (int h = 0; h < 16; ++h)
                bb[h] = lds_kv[bbase + h * HWDIM];
            oacc[j] = __builtin_amdgcn_wmma_f32_16x16x32_f16(false, a, false, bb,
                                                             (short)0, oacc[j], false, false);
        }
        __syncthreads();
    }

    const float g = gamma[0];
    #pragma unroll
    for (int j = 0; j < 9; ++j) {
        const int col = (wave + 4 * j) * 16 + l16;
        #pragma unroll
        for (int r = 0; r < 8; ++r) {
            const int M = r + lhalf * 8;
            const size_t idx = ((size_t)batch * CDIM + m0 + M) * HWDIM + col;
            out[idx] = g * oacc[j][r] * inv_row[M] + feat1[idx];
        }
    }
}

extern "C" void kernel_launch(void* const* d_in, const int* in_sizes, int n_in,
                              void* d_out, int out_size, void* d_ws, size_t ws_size,
                              hipStream_t stream) {
    const float* feat1 = (const float*)d_in[0];
    const float* feat2 = (const float*)d_in[1];
    const float* gamma = (const float*)d_in[2];
    float* out = (float*)d_out;
    (void)in_sizes; (void)n_in; (void)out_size;

    const size_t tensor_bytes = (size_t)NELEM * sizeof(_Float16);   // 28,311,552
    if (ws_size >= 3 * tensor_bytes) {
        _Float16* f1h = (_Float16*)d_ws;
        _Float16* f2h = f1h + NELEM;
        _Float16* f2t = f2h + NELEM;

        convert_f16_kernel<<<dim3(NELEM / 4 / 256), 256, 0, stream>>>(feat1, feat2, f1h, f2h);
        transpose_f16_kernel<<<dim3(CDIM / 32, HWDIM / 32, BATCH), dim3(32, 8), 0, stream>>>(feat2, f2t);
        cma_wmma_kernel<<<dim3(BATCH * (CDIM / 32)), 256, 0, stream>>>(feat1, f1h, f2h, f2t, gamma, out);
    } else {
        cma_fused_kernel<<<dim3(BATCH * (CDIM / 16)), 128, 0, stream>>>(feat1, feat2, gamma, out);
    }
}
